// FusionMamba_7834020347953
// MI455X (gfx1250) — compile-verified, matrix operand order checked by reading, not run on hardware
//
#include <hip/hip_runtime.h>

typedef __attribute__((ext_vector_type(16))) __bf16 v16bf;
typedef __attribute__((ext_vector_type(8)))  __bf16 v8bf;    // 16-byte half-fragment
typedef __attribute__((ext_vector_type(8)))  float  v8f;

#define BATCH  4
#define LSEQ   4096
#define DMODEL 64
#define DIN    64
#define NSTT   8
#define DTR    4
#define KC     4
#define MROWS  (BATCH*LSEQ)   // 16384

static __device__ __forceinline__ float siluf(float x){ return x * (1.f/(1.f + __expf(-x))); }
static __device__ __forceinline__ float softplusf(float x){ return (x > 20.f) ? x : log1pf(__expf(x)); }

// ---------------------------------------------------------------------------
// (B, C, H*W) -> (B, L, C)
__global__ void k_seq_transpose(float* __restrict__ dst, const float* __restrict__ src) {
    int tid = blockIdx.x * blockDim.x + threadIdx.x;
    if (tid >= MROWS * DMODEL) return;
    int c = tid & 63;
    int l = (tid >> 6) & (LSEQ - 1);
    int b = tid >> 18;
    dst[tid] = src[(b * DMODEL + c) * LSEQ + l];
}

// ---------------------------------------------------------------------------
// Pack fp32 weight src[R][Ksrc] (out-features x in-features) into per-lane
// WMMA B fragments (bf16). Fragment layout per 16x16 tile at ktile kt:
//   lane: half=lane>>4, idx16=lane&15;  n = nt*16+idx16
//   elem e<8 : k = kt*32 + half*8 + e          (B[k][n])
//   elem e>=8: k = kt*32 + half*8 + 8 + e
// dst index = ((kt*(N/16)+nt)*32 + lane)*16 + e ;  zero-pad n>=R.
__global__ void k_wpack_bf16(__bf16* __restrict__ dst, const float* __restrict__ src,
                             int R, int Ksrc, int N, int K) {
    int total = (K / 32) * (N / 16) * 512;
    int tid = blockIdx.x * blockDim.x + threadIdx.x;
    if (tid >= total) return;
    int e    = tid & 15;
    int lane = (tid >> 4) & 31;
    int tile = tid >> 9;
    int nt   = tile % (N / 16);
    int kt   = tile / (N / 16);
    int half = lane >> 4, idx16 = lane & 15;
    int k = kt * 32 + half * 8 + ((e < 8) ? e : e + 8);
    int n = nt * 16 + idx16;
    dst[tid] = (n < R && k < Ksrc) ? (__bf16)src[n * Ksrc + k] : (__bf16)0.f;
}

__global__ void k_negexp(float* __restrict__ dst, const float* __restrict__ src, int n) {
    int tid = blockIdx.x * blockDim.x + threadIdx.x;
    if (tid < n) dst[tid] = -__expf(src[tid]);
}

// ---------------------------------------------------------------------------
// row LayerNorm over last dim 64, fp32 in -> bf16 out. one wave per row.
__global__ void k_layernorm_bf16(__bf16* __restrict__ out, const float* __restrict__ x,
                                 const float* __restrict__ w, const float* __restrict__ bia) {
    int row  = blockIdx.x * 8 + (threadIdx.x >> 5);
    int lane = threadIdx.x & 31;
    const float* xr = x + (size_t)row * 64;
    float v0 = xr[lane], v1 = xr[lane + 32];
    float s = v0 + v1, sq = v0 * v0 + v1 * v1;
    #pragma unroll
    for (int m = 16; m >= 1; m >>= 1) { s += __shfl_xor(s, m, 32); sq += __shfl_xor(sq, m, 32); }
    float mean = s * (1.f / 64.f);
    float var  = sq * (1.f / 64.f) - mean * mean;
    float rs   = rsqrtf(var + 1e-5f);
    out[(size_t)row * 64 + lane]      = (__bf16)((v0 - mean) * rs * w[lane]      + bia[lane]);
    out[(size_t)row * 64 + lane + 32] = (__bf16)((v1 - mean) * rs * w[lane + 32] + bia[lane + 32]);
}

// ---------------------------------------------------------------------------
// WMMA bf16 GEMM: C[M,N] = A[M,K] * W (+bias[N]) (+skip[M,N])
// B given fragment-packed (k_wpack_bf16).  One wave per 16x16 tile.
// N, K compile-time -> immediate-offset addressing, branch-free epilogue.
template<int N, int K, bool HAS_BIAS, bool HAS_SKIP>
__global__ __launch_bounds__(32)
void k_gemm_t(const __bf16* __restrict__ A, const __bf16* __restrict__ Bpack,
              float* __restrict__ C, const float* __restrict__ bias,
              const float* __restrict__ skip) {
    constexpr int NT = N / 16;
    int mt = blockIdx.x, nt = blockIdx.y;
    int lane  = threadIdx.x;
    int half  = lane >> 4;
    int idx16 = lane & 15;
    int m0 = mt * 16 + idx16;
    int n0 = nt * 16 + idx16;
    const __bf16* ap = A + m0 * K;
    v8f acc = {};
    #pragma unroll
    for (int kt = 0; kt < K / 32; ++kt) {
        int kb = kt * 32 + half * 8;
        v16bf a, b;
        ((v8bf*)&a)[0] = *(const v8bf*)(ap + kb);        // K kb..kb+7
        ((v8bf*)&a)[1] = *(const v8bf*)(ap + kb + 16);   // K kb+16..kb+23
        const __bf16* bp = Bpack + ((kt * NT + nt) * 32 + lane) * 16;
        ((v8bf*)&b)[0] = *(const v8bf*)(bp);
        ((v8bf*)&b)[1] = *(const v8bf*)(bp + 8);
        acc = __builtin_amdgcn_wmma_f32_16x16x32_bf16(false, a, false, b,
                                                      (short)0, acc, false, false);
    }
    float bv = HAS_BIAS ? bias[n0] : 0.f;
    int mbase = mt * 16 + half * 8;                       // C/D: VGPR j -> M = mbase+j
    float* cp = C + mbase * N + n0;
    const float* sp = HAS_SKIP ? (skip + mbase * N + n0) : nullptr;
    #pragma unroll
    for (int j = 0; j < 8; ++j) {
        float v = acc[j] + bv;
        if (HAS_SKIP) v += sp[j * N];
        cp[j * N] = v;
    }
}

// Fusion GEMM: N=64, transposed store to (B, 64, L), + bias
template<int K>
__global__ __launch_bounds__(32)
void k_gemm_fusion_t(const __bf16* __restrict__ A, const __bf16* __restrict__ Bpack,
                     float* __restrict__ out, const float* __restrict__ bias) {
    constexpr int N = 64, NT = N / 16;
    int mt = blockIdx.x, nt = blockIdx.y;
    int lane  = threadIdx.x;
    int half  = lane >> 4;
    int idx16 = lane & 15;
    int m0 = mt * 16 + idx16;
    int n0 = nt * 16 + idx16;
    const __bf16* ap = A + m0 * K;
    v8f acc = {};
    #pragma unroll
    for (int kt = 0; kt < K / 32; ++kt) {
        int kb = kt * 32 + half * 8;
        v16bf a, b;
        ((v8bf*)&a)[0] = *(const v8bf*)(ap + kb);
        ((v8bf*)&a)[1] = *(const v8bf*)(ap + kb + 16);
        const __bf16* bp = Bpack + ((kt * NT + nt) * 32 + lane) * 16;
        ((v8bf*)&b)[0] = *(const v8bf*)(bp);
        ((v8bf*)&b)[1] = *(const v8bf*)(bp + 8);
        acc = __builtin_amdgcn_wmma_f32_16x16x32_bf16(false, a, false, b,
                                                      (short)0, acc, false, false);
    }
    float bv = bias[n0];
    int mbase = mt * 16 + half * 8;
    int bb = mbase >> 12;                                 // batch
    int lb = mbase & (LSEQ - 1);                          // position
    float* op = out + (bb * 64 + n0) * LSEQ + lb;         // consecutive j -> +1 in L
    #pragma unroll
    for (int j = 0; j < 8; ++j) op[j] = acc[j] + bv;
}

// ---------------------------------------------------------------------------
// depthwise conv (causal fwd, anti-causal bwd == flip->conv->flip) + SiLU
__global__ void k_conv_silu(float* __restrict__ uf, float* __restrict__ ub,
                            const float* __restrict__ xz,
                            const float* __restrict__ wf, const float* __restrict__ bf,
                            const float* __restrict__ wb, const float* __restrict__ bb) {
    int tid = blockIdx.x * blockDim.x + threadIdx.x;
    if (tid >= MROWS * DIN) return;
    int d = tid & 63;
    int r = tid >> 6;
    int l = r & (LSEQ - 1);
    int rb = r - l;                      // batch base row
    float af = bf[d], ab = bb[d];
    #pragma unroll
    for (int j = 0; j < KC; ++j) {
        int lf = l - (KC - 1) + j;
        int lb = l + (KC - 1) - j;
        if (lf >= 0)    af += wf[d * KC + j] * xz[(size_t)(rb + lf) * 128 + d];
        if (lb < LSEQ)  ab += wb[d * KC + j] * xz[(size_t)(rb + lb) * 128 + d];
    }
    uf[tid] = siluf(af);
    ub[tid] = siluf(ab);
}

// dt = softplus(dbl[:, :DTR] @ dtproj^T + b)
__global__ void k_dtproj(float* __restrict__ dtf, float* __restrict__ dtb,
                         const float* __restrict__ dblf, const float* __restrict__ dblb,
                         const float* __restrict__ wtf, const float* __restrict__ btf,
                         const float* __restrict__ wtb, const float* __restrict__ btb) {
    int tid = blockIdx.x * blockDim.x + threadIdx.x;
    if (tid >= MROWS * DIN) return;
    int dir = blockIdx.y;
    const float* dbl = dir ? dblb : dblf;
    const float* wt  = dir ? wtb  : wtf;
    const float* bt  = dir ? btb  : btf;
    float*       dt  = dir ? dtb  : dtf;
    int d = tid & 63;
    int r = tid >> 6;
    float s = bt[d];
    #pragma unroll
    for (int j = 0; j < DTR; ++j) s += dbl[(size_t)r * 32 + j] * wt[d * DTR + j];
    dt[tid] = softplusf(s);
}

// ---------------------------------------------------------------------------
// selective scan. grid=(BATCH, 2 dirs), block=512 (=16 waves).
// thread = (d,n): holds state h[d][n]; reduce over n (8 lanes) via shfl_xor.
__global__ __launch_bounds__(512)
void k_scan(float* __restrict__ y2,
            const float* __restrict__ dtf, const float* __restrict__ dtb,
            const float* __restrict__ uf,  const float* __restrict__ ub,
            const float* __restrict__ dblf, const float* __restrict__ dblb,
            const float* __restrict__ Af,  const float* __restrict__ Ab,
            const float* __restrict__ Df,  const float* __restrict__ Db) {
    int b   = blockIdx.x;
    int dir = blockIdx.y;
    int d = threadIdx.x >> 3;
    int n = threadIdx.x & 7;
    const float* DT  = dir ? dtb  : dtf;
    const float* U   = dir ? ub   : uf;
    const float* DBL = dir ? dblb : dblf;
    float A  = (dir ? Ab : Af)[d * NSTT + n];
    float Dd = (dir ? Db : Df)[d];
    float* Y = y2 + (size_t)dir * MROWS * DIN;
    int l0 = dir ? (LSEQ - 1) : 0;
    int dl = dir ? -1 : 1;
    float h = 0.f;
    int base = b * LSEQ;
    for (int s = 0; s < LSEQ; ++s) {
        int l = l0 + dl * s;
        size_t r = (size_t)(base + l);
        float dt = DT[r * DIN + d];
        float u  = U [r * DIN + d];
        float Bv = DBL[r * 32 + DTR + n];
        float Cv = DBL[r * 32 + DTR + NSTT + n];
        if (s + 1 < LSEQ) {
            size_t rn = (size_t)(base + l + dl);
            __builtin_prefetch(&DT[rn * DIN + d], 0, 1);
            __builtin_prefetch(&U [rn * DIN + d], 0, 1);
            __builtin_prefetch(&DBL[rn * 32 + n], 0, 1);
        }
        h = __expf(dt * A) * h + dt * u * Bv;
        float p = h * Cv;
        p += __shfl_xor(p, 1, 32);
        p += __shfl_xor(p, 2, 32);
        p += __shfl_xor(p, 4, 32);
        if (n == 0) Y[r * DIN + d] = p + u * Dd;
    }
}

// y = 0.5*(y_f+y_b); LN; * silu(z) -> bf16. one wave per row.
__global__ void k_combine(__bf16* __restrict__ g, const float* __restrict__ y2,
                          const float* __restrict__ xz,
                          const float* __restrict__ lnw, const float* __restrict__ lnb) {
    int row  = blockIdx.x * 8 + (threadIdx.x >> 5);
    int lane = threadIdx.x & 31;
    const float* y0 = y2;
    const float* y1 = y2 + (size_t)MROWS * DIN;
    size_t r64 = (size_t)row * 64;
    float v0 = 0.5f * (y0[r64 + lane]      + y1[r64 + lane]);
    float v1 = 0.5f * (y0[r64 + lane + 32] + y1[r64 + lane + 32]);
    float s = v0 + v1, sq = v0 * v0 + v1 * v1;
    #pragma unroll
    for (int m = 16; m >= 1; m >>= 1) { s += __shfl_xor(s, m, 32); sq += __shfl_xor(sq, m, 32); }
    float mean = s * (1.f / 64.f);
    float var  = sq * (1.f / 64.f) - mean * mean;
    float rs   = rsqrtf(var + 1e-5f);
    size_t r128 = (size_t)row * 128;
    float z0 = xz[r128 + 64 + lane], z1 = xz[r128 + 96 + lane];
    g[r64 + lane]      = (__bf16)(((v0 - mean) * rs * lnw[lane]      + lnb[lane])      * siluf(z0));
    g[r64 + lane + 32] = (__bf16)(((v1 - mean) * rs * lnw[lane + 32] + lnb[lane + 32]) * siluf(z1));
}

// avg of concatenated pairs -> bf16 (B,L,128)
__global__ void k_avg_bf16(__bf16* __restrict__ avg,
                           const float* __restrict__ i1l, const float* __restrict__ i1g,
                           const float* __restrict__ i2l, const float* __restrict__ i2g) {
    int tid = blockIdx.x * blockDim.x + threadIdx.x;
    if (tid >= MROWS * 128) return;
    int e = tid & 127;
    int r = tid >> 7;
    float v;
    if (e < 64) v = 0.5f * (i1l[(size_t)r * 64 + e]        + i2l[(size_t)r * 64 + e]);
    else        v = 0.5f * (i1g[(size_t)r * 64 + (e - 64)] + i2g[(size_t)r * 64 + (e - 64)]);
    avg[tid] = (__bf16)v;
}

// ---------------------------------------------------------------------------
extern "C" void kernel_launch(void* const* d_in, const int* in_sizes, int n_in,
                              void* d_out, int out_size, void* d_ws, size_t ws_size,
                              hipStream_t stream) {
    (void)in_sizes; (void)n_in; (void)out_size; (void)ws_size;
    const float* img[4]   = {(const float*)d_in[0], (const float*)d_in[1],
                             (const float*)d_in[2], (const float*)d_in[3]};
    const float* norm0_w  = (const float*)d_in[4];
    const float* norm0_b  = (const float*)d_in[5];
    const float* norm1_w  = (const float*)d_in[6];
    const float* norm1_b  = (const float*)d_in[7];
    const float* in_w     = (const float*)d_in[8];
    const float* in_b     = (const float*)d_in[9];
    const float* conv_w_f = (const float*)d_in[10];
    const float* conv_b_f = (const float*)d_in[11];
    const float* conv_w_b = (const float*)d_in[12];
    const float* conv_b_b = (const float*)d_in[13];
    const float* xproj_f  = (const float*)d_in[14];
    const float* xproj_b  = (const float*)d_in[15];
    const float* dtw_f    = (const float*)d_in[16];
    const float* dtb_f    = (const float*)d_in[17];
    const float* dtw_b    = (const float*)d_in[18];
    const float* dtb_b    = (const float*)d_in[19];
    const float* Alog_f   = (const float*)d_in[20];
    const float* Alog_b   = (const float*)d_in[21];
    const float* D_f      = (const float*)d_in[22];
    const float* D_b      = (const float*)d_in[23];
    const float* ln_w     = (const float*)d_in[24];
    const float* ln_b     = (const float*)d_in[25];
    const float* mout_w   = (const float*)d_in[26];
    const float* mout_b   = (const float*)d_in[27];
    const float* fop_w    = (const float*)d_in[28];
    const float* fop_b    = (const float*)d_in[29];

    char* w = (char*)d_ws;
    const size_t MB = 1024ull * 1024ull;
    float*  SEQ[4];  for (int i = 0; i < 4; ++i) SEQ[i]  = (float*)(w + (size_t)i * 4 * MB);
    float*  NSEQ[4]; for (int i = 0; i < 4; ++i) NSEQ[i] = (float*)(w + (16 + (size_t)i * 4) * MB);
    float*  IB[4];   for (int i = 0; i < 4; ++i) IB[i]   = (float*)(w + (32 + (size_t)i * 4) * MB);
    float*  XZ   = (float*)(w + 48 * MB);           // 8 MB
    float*  UF   = (float*)(w + 56 * MB);
    float*  UB   = (float*)(w + 60 * MB);
    float*  DTF  = (float*)(w + 64 * MB);
    float*  DTB  = (float*)(w + 68 * MB);
    float*  Y2   = (float*)(w + 72 * MB);           // 8 MB (2 dirs)
    float*  DBLF = (float*)(w + 80 * MB);           // 2 MB (padded to 32)
    float*  DBLB = (float*)(w + 82 * MB);
    __bf16* XLN  = (__bf16*)(w + 84 * MB);
    __bf16* ELN  = (__bf16*)(w + 86 * MB);
    __bf16* G    = (__bf16*)(w + 88 * MB);
    __bf16* AVG  = (__bf16*)(w + 48 * MB);          // reuse XZ region after block loop
    __bf16* pIn  = (__bf16*)(w + 90 * MB);                 // 2*8*512  elems
    __bf16* pXf  = (__bf16*)(w + 90 * MB + 32 * 1024);     // 2*2*512
    __bf16* pXb  = (__bf16*)(w + 90 * MB + 64 * 1024);
    __bf16* pM   = (__bf16*)(w + 90 * MB + 96 * 1024);     // 2*4*512
    __bf16* pFop = (__bf16*)(w + 90 * MB + 128 * 1024);    // 4*4*512
    float*  Af   = (float*)(w + 90 * MB + 160 * 1024);
    float*  Ab   = (float*)(w + 90 * MB + 164 * 1024);

    // sequence-ify the four images
    for (int i = 0; i < 4; ++i)
        k_seq_transpose<<<(MROWS * DMODEL) / 256, 256, 0, stream>>>(SEQ[i], img[i]);

    // fop_w (64 x 128) -> packed fragments (N=64, K=128)
    k_wpack_bf16<<<(4 * 4 * 512) / 256, 256, 0, stream>>>(pFop, fop_w, 64, 128, 64, 128);

    // block wiring: {x, extra, dst}
    const float* xs[8] = {SEQ[0], SEQ[1], SEQ[2], SEQ[3], NSEQ[0], NSEQ[0], NSEQ[1], NSEQ[1]};
    const float* es[8] = {SEQ[2], SEQ[3], NSEQ[0], NSEQ[1], NSEQ[1], NSEQ[3], NSEQ[0], NSEQ[2]};
    float*       ds[8] = {NSEQ[0], NSEQ[1], NSEQ[2], NSEQ[3], IB[0], IB[1], IB[2], IB[3]};

    for (int i = 0; i < 8; ++i) {
        // per-block weight prep (fragment-packed bf16)
        k_wpack_bf16<<<(2 * 8 * 512) / 256, 256, 0, stream>>>(pIn, in_w + (size_t)i * 128 * 64, 128, 64, 128, 64);
        k_wpack_bf16<<<(2 * 2 * 512) / 256, 256, 0, stream>>>(pXf, xproj_f + (size_t)i * 20 * 64, 20, 64, 32, 64);
        k_wpack_bf16<<<(2 * 2 * 512) / 256, 256, 0, stream>>>(pXb, xproj_b + (size_t)i * 20 * 64, 20, 64, 32, 64);
        k_wpack_bf16<<<(2 * 4 * 512) / 256, 256, 0, stream>>>(pM, mout_w + (size_t)i * 64 * 64, 64, 64, 64, 64);
        k_negexp<<<2, 256, 0, stream>>>(Af, Alog_f + (size_t)i * DIN * NSTT, DIN * NSTT);
        k_negexp<<<2, 256, 0, stream>>>(Ab, Alog_b + (size_t)i * DIN * NSTT, DIN * NSTT);

        // LayerNorms -> bf16
        k_layernorm_bf16<<<MROWS / 8, 256, 0, stream>>>(XLN, xs[i], norm0_w + i * 64, norm0_b + i * 64);
        k_layernorm_bf16<<<MROWS / 8, 256, 0, stream>>>(ELN, es[i], norm1_w + i * 64, norm1_b + i * 64);

        // input projection (WMMA): xz = LN(x) @ in_w^T + b  -> (M,128)
        k_gemm_t<128, 64, true, false><<<dim3(MROWS / 16, 8), 32, 0, stream>>>(
            XLN, pIn, XZ, in_b + (size_t)i * 128, nullptr);

        // depthwise conv + silu (both directions, flip-free)
        k_conv_silu<<<(MROWS * DIN) / 256, 256, 0, stream>>>(UF, UB, XZ,
            conv_w_f + (size_t)i * DIN * KC, conv_b_f + i * DIN,
            conv_w_b + (size_t)i * DIN * KC, conv_b_b + i * DIN);

        // x-projection (WMMA, N padded 20->32): dbl = LN(extra) @ xproj^T
        k_gemm_t<32, 64, false, false><<<dim3(MROWS / 16, 2), 32, 0, stream>>>(
            ELN, pXf, DBLF, nullptr, nullptr);
        k_gemm_t<32, 64, false, false><<<dim3(MROWS / 16, 2), 32, 0, stream>>>(
            ELN, pXb, DBLB, nullptr, nullptr);

        // dt projection + softplus
        k_dtproj<<<dim3((MROWS * DIN) / 256, 2), 256, 0, stream>>>(DTF, DTB, DBLF, DBLB,
            dtw_f + (size_t)i * DIN * DTR, dtb_f + i * DIN,
            dtw_b + (size_t)i * DIN * DTR, dtb_b + i * DIN);

        // selective scan (fwd ascending, bwd descending)
        k_scan<<<dim3(BATCH, 2), 512, 0, stream>>>(Y2, DTF, DTB, UF, UB, DBLF, DBLB,
            Af, Ab, D_f + i * DIN, D_b + i * DIN);

        // combine dirs, LN, gate -> bf16
        k_combine<<<MROWS / 8, 256, 0, stream>>>(G, Y2, XZ, ln_w + i * DIN, ln_b + i * DIN);

        // output projection + skip (WMMA)
        k_gemm_t<64, 64, true, true><<<dim3(MROWS / 16, 4), 32, 0, stream>>>(
            G, pM, ds[i], mout_b + (size_t)i * DMODEL, xs[i]);
    }

    // fusion: avg of concatenated pairs -> bf16, WMMA with transposed store
    k_avg_bf16<<<(MROWS * 128) / 256, 256, 0, stream>>>(AVG, IB[0], IB[1], IB[2], IB[3]);
    k_gemm_fusion_t<128><<<dim3(MROWS / 16, 4), 32, 0, stream>>>(AVG, pFop, (float*)d_out, fop_b);
}
